// SympleAgent_32255204393220
// MI455X (gfx1250) — matrix operand order repaired and checked by reading, not compiled, not run on hardware
//
#include <hip/hip_runtime.h>
#include <math.h>

typedef __attribute__((ext_vector_type(16))) _Float16 v16h;
typedef __attribute__((ext_vector_type(8)))  _Float16 v8h;
typedef __attribute__((ext_vector_type(8)))  float    v8f;
typedef __attribute__((ext_vector_type(4)))  float    v4f;

#define N_NODES 400000
#define N_TILES 25000   // 16 nodes per tile
#define OUT_HEAD 39     // 4 + 5 + 30

// ---------------------------------------------------------------------------
// Kernel 1: the three tiny heads on row 0 (negligible work, 1 block)
// ---------------------------------------------------------------------------
__device__ __forceinline__ void run_head(const float* feats, float* hidden, int t,
                                         const float* W1, const float* b1,
                                         const float* W2, const float* b2,
                                         int nout, float* outp, const float* validity)
{
    if (t < 128) {
        float s = b1[t];
        #pragma unroll 8
        for (int k = 0; k < 264; ++k) s = fmaf(feats[k], W1[k * 128 + t], s);
        hidden[t] = fmaxf(s, 0.0f);
    }
    __syncthreads();
    if (t < nout) {
        float s = b2[t];
        #pragma unroll 8
        for (int j = 0; j < 128; ++j) s = fmaf(hidden[j], W2[j * nout + t], s);
        if (validity) s += logf(validity[t]);
        outp[t] = s;
    }
    __syncthreads();
}

__global__ void __launch_bounds__(256)
heads_kernel(const float* __restrict__ node_hidden, const float* __restrict__ h_glob,
             const float* __restrict__ state, const float* __restrict__ validity,
             const float* __restrict__ W_hl1, const float* __restrict__ b_hl1,
             const float* __restrict__ W_hl2, const float* __restrict__ b_hl2,
             const float* __restrict__ W_in1, const float* __restrict__ b_in1,
             const float* __restrict__ W_in2, const float* __restrict__ b_in2,
             const float* __restrict__ W_ex1, const float* __restrict__ b_ex1,
             const float* __restrict__ W_ex2, const float* __restrict__ b_ex2,
             float* __restrict__ out)
{
    __shared__ float feats[264];
    __shared__ float hidden[128];
    int t = threadIdx.x;
    if (t < 128)                feats[t]       = node_hidden[t];   // cur = row 0
    else if (t < 256)           feats[t]       = h_glob[t - 128];
    if (t < 8)                  feats[256 + t] = state[t];
    __syncthreads();
    run_head(feats, hidden, t, W_hl1, b_hl1, W_hl2, b_hl2,  4, out + 0, nullptr);
    run_head(feats, hidden, t, W_in1, b_in1, W_in2, b_in2,  5, out + 4, nullptr);
    run_head(feats, hidden, t, W_ex1, b_ex1, W_ex2, b_ex2, 30, out + 9, validity);
}

// ---------------------------------------------------------------------------
// Kernel 2: teleport logits via f16 WMMA.  H^T = (W_mid)^T x (node)^T
//   A (16x32 f16) : weight tile in LDS (contiguous-K lane layout), streamed per use
//   B (32x16 f16) : 16 nodes' features, lane(&15)=node, 16 contiguous K per lane
//   C (16x16 f32) : fused bias fragment initializes the accumulator
//   D (16x16 f32) : lane&15 = node, vgpr r -> hidden unit r + 8*(lane>>4)
// An opaque zero offset (volatile inline asm) blocks LICM from hoisting the
// LDS reads out of the tile loop, keeping VGPR pressure ~100 -> high occupancy.
// ---------------------------------------------------------------------------
__global__ void __launch_bounds__(256)
teleport_kernel(const float* __restrict__ node_hidden,
                const float* __restrict__ h_glob, const float* __restrict__ state,
                const float* __restrict__ W_tp1, const float* __restrict__ b_tp1,
                const float* __restrict__ W_tp2, const float* __restrict__ b_tp2,
                float* __restrict__ out)
{
    __shared__ _Float16 sA[4 * 8 * 32 * 16];   // [kb][mtile][lane][e]  = 32 KB
    __shared__ float    sC[128];               // fused bias: b_tp1 + hglob/state part
    __shared__ float    sW2[128];

    const int t = threadIdx.x;

    // ---- pre-pack A fragments (constant weights) in contiguous-K 16-bit A layout ----
    // lane 0-15: K = e;  lane 16-31: K = 16 + e.   A[m][k] = W_tp1[128+k][m]
    for (int idx = t; idx < 16384; idx += 256) {
        int e  = idx & 15;
        int ln = (idx >> 4) & 31;
        int mt = (idx >> 9) & 7;
        int kb = idx >> 12;
        int kl = e + ((ln >> 4) << 4);
        int k  = kb * 32 + kl;
        int m  = mt * 16 + (ln & 15);
        sA[idx] = (_Float16)W_tp1[(size_t)(128 + k) * 128 + m];
    }
    // ---- fused constant bias c[j] and second-layer weights ----
    if (t < 128) {
        float s = b_tp1[t];
        #pragma unroll 8
        for (int k = 0; k < 128; ++k) s = fmaf(h_glob[k], W_tp1[k * 128 + t], s);
        #pragma unroll
        for (int k = 0; k < 8;   ++k) s = fmaf(state[k],  W_tp1[(256 + k) * 128 + t], s);
        sC[t]  = s;
        sW2[t] = W_tp2[t];
    }
    __syncthreads();

    const int lane   = t & 31;
    const int warp   = t >> 5;
    const int half   = lane >> 4;     // selects K-subset for A/B, M-subset for C/D
    const int nIdx   = lane & 15;     // node index within tile (B column / D column)
    const int wavesT = gridDim.x * (blockDim.x >> 5);
    const int waveId = blockIdx.x * (blockDim.x >> 5) + warp;
    const float bias2 = b_tp2[0];

    for (int tile = waveId; tile < N_TILES; tile += wavesT) {
        // Opaque zero: defeats LICM so per-tile LDS reads stay inside the loop.
        int zoff;
        asm volatile("s_mov_b32 %0, 0" : "=s"(zoff));
        const _Float16* sAo  = sA  + zoff;
        const float*    sCo  = sC  + zoff;
        const float*    sW2o = sW2 + zoff;

        const float* nodeBase = node_hidden + ((size_t)(tile * 16 + nIdx)) * 128;

        // prefetch 2 tiles ahead; the two half-wave groups split the 4 cache
        // lines of each 512B row so the whole 8KB tile is covered by 2 insts.
        int nt = tile + 2 * wavesT;
        if (nt < N_TILES) {
            const float* pf = node_hidden + ((size_t)(nt * 16 + nIdx)) * 128;
            __builtin_prefetch(pf + (half * 2 + 0) * 32, 0, 0);
            __builtin_prefetch(pf + (half * 2 + 1) * 32, 0, 0);
        }

        // ---- load & convert B fragments: lane reads 16 contiguous f32 per kb ----
        v16h B[4];
        #pragma unroll
        for (int kb = 0; kb < 4; ++kb) {
            const v4f* p = (const v4f*)(nodeBase + kb * 32 + half * 16);
            float fb[16];
            *(v4f*)(fb + 0)  = p[0];
            *(v4f*)(fb + 4)  = p[1];
            *(v4f*)(fb + 8)  = p[2];
            *(v4f*)(fb + 12) = p[3];
            v16h b;
            #pragma unroll
            for (int e = 0; e < 16; ++e) b[e] = (_Float16)fb[e];
            B[kb] = b;
        }

        // ---- 8 m-tiles x 4 k-steps of WMMA; bias enters via the C operand ----
        float partial = 0.0f;
        #pragma unroll
        for (int mt = 0; mt < 8; ++mt) {
            // C fragment: acc[r] starts at c[mt*16 + 8*half + r]
            const float* cp = sCo + half * 8 + mt * 16;
            v4f c0 = *(const v4f*)(cp + 0);
            v4f c1 = *(const v4f*)(cp + 4);
            v8f acc = __builtin_shufflevector(c0, c1, 0, 1, 2, 3, 4, 5, 6, 7);
            #pragma unroll
            for (int kb = 0; kb < 4; ++kb) {
                const v8h* ap = (const v8h*)(sAo + (((kb * 8 + mt) * 32 + lane) << 4));
                v8h lo = ap[0], hi = ap[1];
                v16h a = __builtin_shufflevector(lo, hi,
                        0, 1, 2, 3, 4, 5, 6, 7, 8, 9, 10, 11, 12, 13, 14, 15);
                acc = __builtin_amdgcn_wmma_f32_16x16x32_f16(
                        false, a, false, B[kb], (short)0, acc, false, false);
            }
            const float* wp = sW2o + half * 8 + mt * 16;
            float wb[8];
            *(v4f*)(wb + 0) = *(const v4f*)(wp + 0);
            *(v4f*)(wb + 4) = *(const v4f*)(wp + 4);
            #pragma unroll
            for (int r = 0; r < 8; ++r) {
                float h = acc[r] < 0.0f ? 0.0f : acc[r];   // ReLU, single op
                partial = fmaf(h, wb[r], partial);
            }
        }

        // combine the two half-wave partial sums (hidden units split across halves)
        partial += __shfl_xor(partial, 16, 32);
        if (lane < 16)
            out[OUT_HEAD + tile * 16 + lane] = partial + bias2;
    }
}

// ---------------------------------------------------------------------------
extern "C" void kernel_launch(void* const* d_in, const int* in_sizes, int n_in,
                              void* d_out, int out_size, void* d_ws, size_t ws_size,
                              hipStream_t stream)
{
    (void)in_sizes; (void)n_in; (void)d_ws; (void)ws_size; (void)out_size;
    const float* node_hidden = (const float*)d_in[0];
    const float* h_glob      = (const float*)d_in[1];
    const float* state       = (const float*)d_in[2];
    const float* validity    = (const float*)d_in[3];
    const float* W_hl1 = (const float*)d_in[4];   const float* b_hl1 = (const float*)d_in[5];
    const float* W_hl2 = (const float*)d_in[6];   const float* b_hl2 = (const float*)d_in[7];
    const float* W_in1 = (const float*)d_in[8];   const float* b_in1 = (const float*)d_in[9];
    const float* W_in2 = (const float*)d_in[10];  const float* b_in2 = (const float*)d_in[11];
    const float* W_ex1 = (const float*)d_in[12];  const float* b_ex1 = (const float*)d_in[13];
    const float* W_ex2 = (const float*)d_in[14];  const float* b_ex2 = (const float*)d_in[15];
    const float* W_tp1 = (const float*)d_in[16];  const float* b_tp1 = (const float*)d_in[17];
    const float* W_tp2 = (const float*)d_in[18];  const float* b_tp2 = (const float*)d_in[19];
    float* out = (float*)d_out;

    heads_kernel<<<1, 256, 0, stream>>>(node_hidden, h_glob, state, validity,
                                        W_hl1, b_hl1, W_hl2, b_hl2,
                                        W_in1, b_in1, W_in2, b_in2,
                                        W_ex1, b_ex1, W_ex2, b_ex2, out);

    teleport_kernel<<<782, 256, 0, stream>>>(node_hidden, h_glob, state,
                                             W_tp1, b_tp1, W_tp2, b_tp2, out);
}
